// ATTNRNNAgent_14757507629754
// MI455X (gfx1250) — matrix operand order repaired and checked by reading, not compiled, not run on hardware
//
#include <hip/hip_runtime.h>
#include <hip/hip_bf16.h>

typedef __bf16 bf16;
typedef __attribute__((ext_vector_type(16))) __bf16 v16bf;
typedef __attribute__((ext_vector_type(8)))  __bf16 v8bf;
typedef __attribute__((ext_vector_type(8)))  float   v8f;

#define B_TOT 4096
#define NN 64
#define EE 128
#define AA 32
#define HID 256
#define NACT 32
#define NEGV -1000000000.0f

// padded LDS strides (elements)
#define SE 136   // 128 + 8
#define SH 264   // 256 + 8
#define SV 72    // 64 + 8   (v transposed: 256 rows x 64 cols)
#define SW 72
#define SM 72

// LDS pool offsets (bytes), phase-overlaid
#define OFF_SLOTA 0        // s_in (64*136*2=17408)   / s_h0 (32*264*2=16896)
#define OFF_SLOTB 17408    // s_x  (64*264*2=33792)   / s_w  (8*32*72*2=36864)
#define OFF_SLOTC 54272    // s_q  (32*264*2=16896)   / s_h  (16896)
#define OFF_K     71168    // s_k  (33792)
#define OFF_VT    104960   // s_vT (36864)
#define OFF_ATT   141824   // s_att(16896)
#define OFF_MASK  158720   // s_mask (32*72=2304)
#define LDS_BYTES 161024

// bf16 weight image offsets in d_ws (elements)
#define WS_FC1 0
#define WS_Q   32768
#define WS_K   98304
#define WS_V   163840
#define WS_WIH 229376
#define WS_WHH 425984
#define WS_OUT 622592
#define WS_TOTAL 630784

__device__ __forceinline__ v8f wmma_bf16(v16bf a, v16bf b, v8f c) {
  return __builtin_amdgcn_wmma_f32_16x16x32_bf16(false, a, false, b, (short)0, c, false, false);
}

// A-matrix 16x32 bf16 fragment (ISA 7.12.2): lane holds row (lane&15);
// elems 0..7 = K[kb+8*half .. +7], elems 8..15 = K[kb+16+8*half .. +7]
__device__ __forceinline__ v16bf load_a(const bf16* base, int stride, int kbase, int lane) {
  const bf16* p = base + (lane & 15) * stride + kbase + ((lane >> 4) << 3);
  v8bf lo = *reinterpret_cast<const v8bf*>(p);
  v8bf hi = *reinterpret_cast<const v8bf*>(p + 16);
  v16bf r;
#pragma unroll
  for (int i = 0; i < 8; ++i) { r[i] = lo[i]; r[i + 8] = hi[i]; }
  return r;
}

// B-matrix 32x16 bf16 fragment: lane holds column (lane&15);
// elems 0..7 = K[kb+8*half .. +7], elems 8..15 = K[kb+16+8*half .. +7]
__device__ __forceinline__ v16bf load_b(const bf16* base, int stride, int kbase, int lane) {
  const bf16* p = base + (lane & 15) * stride + kbase + ((lane >> 4) << 3);
  v8bf lo = *reinterpret_cast<const v8bf*>(p);
  v8bf hi = *reinterpret_cast<const v8bf*>(p + 16);
  v16bf r;
#pragma unroll
  for (int i = 0; i < 8; ++i) { r[i] = lo[i]; r[i + 8] = hi[i]; }
  return r;
}

__device__ __forceinline__ v8f zero8() {
  v8f z = {0.f, 0.f, 0.f, 0.f, 0.f, 0.f, 0.f, 0.f};
  return z;
}
__device__ __forceinline__ float sigmoidf_(float x) { return 1.0f / (1.0f + __expf(-x)); }

__global__ void convert_weights_kernel(const float* fc1, const float* qw, const float* kw,
                                       const float* vw, const float* wih, const float* whh,
                                       const float* ow, bf16* dst) {
  int i = blockIdx.x * 256 + threadIdx.x;
  if (i >= WS_TOTAL) return;
  float v;
  if      (i < WS_Q)   v = fc1[i - WS_FC1];
  else if (i < WS_K)   v = qw [i - WS_Q];
  else if (i < WS_V)   v = kw [i - WS_K];
  else if (i < WS_WIH) v = vw [i - WS_V];
  else if (i < WS_WHH) v = wih[i - WS_WIH];
  else if (i < WS_OUT) v = whh[i - WS_WHH];
  else                 v = ow [i - WS_OUT];
  dst[i] = (bf16)v;
}

__global__ __launch_bounds__(256) void attn_rnn_agent_kernel(
    const float* __restrict__ g_inputs, const float* __restrict__ g_hidden,
    const float* __restrict__ fc1_b, const float* __restrict__ q_b,
    const float* __restrict__ k_b, const float* __restrict__ v_b,
    const float* __restrict__ b_ih, const float* __restrict__ b_hh,
    const float* __restrict__ out_b,
    const unsigned char* __restrict__ g_obsmask, const unsigned char* __restrict__ g_scen,
    const bf16* __restrict__ ws, float* __restrict__ g_out) {
  __shared__ __align__(16) unsigned char lds[LDS_BYTES];
  bf16* s_in  = (bf16*)(lds + OFF_SLOTA);
  bf16* s_h0  = (bf16*)(lds + OFF_SLOTA);   // overlay (s_in dead after phase 1)
  bf16* s_x   = (bf16*)(lds + OFF_SLOTB);
  bf16* s_w   = (bf16*)(lds + OFF_SLOTB);   // overlay (s_x dead after phase 2)
  bf16* s_q   = (bf16*)(lds + OFF_SLOTC);
  bf16* s_h   = (bf16*)(lds + OFF_SLOTC);   // overlay (s_q dead after scores)
  bf16* s_k   = (bf16*)(lds + OFF_K);
  bf16* s_vT  = (bf16*)(lds + OFF_VT);
  bf16* s_att = (bf16*)(lds + OFF_ATT);
  unsigned char* s_mask = lds + OFF_MASK;

  const int tid   = threadIdx.x;
  const int lane  = tid & 31;
  const int wave  = tid >> 5;
  const int b     = blockIdx.x;
  const int colb  = lane & 15;
  const int halfb = (lane >> 4) << 3;

  const bf16* wfc1 = ws + WS_FC1;
  const bf16* wqm  = ws + WS_Q;
  const bf16* wkm  = ws + WS_K;
  const bf16* wvm  = ws + WS_V;
  const bf16* wih  = ws + WS_WIH;
  const bf16* whh  = ws + WS_WHH;
  const bf16* wout = ws + WS_OUT;

  // ---------- Phase 0: stage inputs + obs_mask into LDS ----------
  {
    const float* gi = g_inputs + (size_t)b * NN * EE;
    for (int i = tid * 4; i < NN * EE; i += 256 * 4) {
      float4 f = *(const float4*)(gi + i);
      int r = i >> 7, c = i & 127;
      bf16* d = s_in + r * SE + c;
      d[0] = (bf16)f.x; d[1] = (bf16)f.y; d[2] = (bf16)f.z; d[3] = (bf16)f.w;
    }
    const unsigned char* gm = g_obsmask + (size_t)b * NN * NN;
    for (int i = tid; i < AA * NN; i += 256)
      s_mask[(i >> 6) * SM + (i & 63)] = gm[i];
  }
  __syncthreads();

  // ---------- Phase 1: x = relu(inputs @ fc1_w^T + b), 64x256, K=128 ----------
  for (int t = wave; t < 64; t += 8) {
    int tm = t >> 4, tn = t & 15;
    v8f acc = zero8();
    const bf16* ab = s_in + tm * 16 * SE;
    const bf16* bb = wfc1 + tn * 16 * EE;
#pragma unroll
    for (int kc = 0; kc < 4; ++kc)
      acc = wmma_bf16(load_a(ab, SE, kc * 32, lane), load_b(bb, EE, kc * 32, lane), acc);
    int col = tn * 16 + colb;
    float bv = fc1_b[col];
    int rb = tm * 16 + halfb;
#pragma unroll
    for (int j = 0; j < 8; ++j) {
      float vv = fmaxf(acc[j] + bv, 0.f);
      s_x[(rb + j) * SH + col] = (bf16)vv;
    }
  }
  __syncthreads();

  // ---------- Phase 2: q,k,v projections (K=256) + stage h0 bf16 ----------
  {
    const float* gh = g_hidden + (size_t)b * AA * HID;
    for (int i = tid * 4; i < AA * HID; i += 256 * 4) {
      float4 f = *(const float4*)(gh + i);
      int r = i >> 8, c = i & 255;
      bf16* d = s_h0 + r * SH + c;
      d[0] = (bf16)f.x; d[1] = (bf16)f.y; d[2] = (bf16)f.z; d[3] = (bf16)f.w;
    }
    for (int t = wave; t < 160; t += 8) {
      const bf16* wb; const float* bias; int tm, tn, mode; // 0=q,1=k,2=v(transposed)
      if (t < 32)      { tm = t >> 4;        tn = t & 15;        wb = wqm; bias = q_b; mode = 0; }
      else if (t < 96) { int u = t - 32; tm = u >> 4; tn = u & 15; wb = wkm; bias = k_b; mode = 1; }
      else             { int u = t - 96; tm = u >> 4; tn = u & 15; wb = wvm; bias = v_b; mode = 2; }
      v8f acc = zero8();
      const bf16* ab = s_x + tm * 16 * SH;
      const bf16* bb = wb + tn * 16 * HID;
#pragma unroll
      for (int kc = 0; kc < 8; ++kc)
        acc = wmma_bf16(load_a(ab, SH, kc * 32, lane), load_b(bb, HID, kc * 32, lane), acc);
      int col = tn * 16 + colb;
      float bv = bias[col];
      int rb = tm * 16 + halfb;
#pragma unroll
      for (int j = 0; j < 8; ++j) {
        float vv = acc[j] + bv;
        if (mode == 2)      { vv = fmaxf(vv, 0.f); s_vT[col * SV + rb + j] = (bf16)vv; }
        else if (mode == 1) s_k[(rb + j) * SH + col] = (bf16)vv;
        else                s_q[(rb + j) * SH + col] = (bf16)vv;
      }
    }
  }
  __syncthreads();

  // ---------- Phase 3: attention, one wave per head ----------
  {
    const int hh = wave;
    v8f sc[2][4];
#pragma unroll
    for (int tm = 0; tm < 2; ++tm)
#pragma unroll
      for (int tn = 0; tn < 4; ++tn)
        sc[tm][tn] = wmma_bf16(load_a(s_q + tm * 16 * SH, SH, hh * 32, lane),
                               load_b(s_k + tn * 16 * SH, SH, hh * 32, lane), zero8());
    const float scale = 0.1767766952966369f; // 1/sqrt(32)
#pragma unroll
    for (int tm = 0; tm < 2; ++tm)
#pragma unroll
      for (int tn = 0; tn < 4; ++tn)
#pragma unroll
        for (int j = 0; j < 8; ++j) {
          int a = tm * 16 + j + halfb, n = tn * 16 + colb;
          float v = sc[tm][tn][j] * scale;
          if (s_mask[a * SM + n]) v = NEGV;
          sc[tm][tn][j] = v;
        }
    // row softmax: row a lives in VGPR j of one 16-lane half
#pragma unroll
    for (int tm = 0; tm < 2; ++tm)
#pragma unroll
      for (int j = 0; j < 8; ++j) {
        float m = fmaxf(fmaxf(sc[tm][0][j], sc[tm][1][j]), fmaxf(sc[tm][2][j], sc[tm][3][j]));
#pragma unroll
        for (int off = 1; off < 16; off <<= 1) m = fmaxf(m, __shfl_xor(m, off, 32));
        float e0 = __expf(sc[tm][0][j] - m), e1 = __expf(sc[tm][1][j] - m);
        float e2 = __expf(sc[tm][2][j] - m), e3 = __expf(sc[tm][3][j] - m);
        float s = e0 + e1 + e2 + e3;
#pragma unroll
        for (int off = 1; off < 16; off <<= 1) s += __shfl_xor(s, off, 32);
        float inv = 1.0f / s;
        sc[tm][0][j] = e0 * inv; sc[tm][1][j] = e1 * inv;
        sc[tm][2][j] = e2 * inv; sc[tm][3][j] = e3 * inv;
      }
#pragma unroll
    for (int tm = 0; tm < 2; ++tm)
#pragma unroll
      for (int tn = 0; tn < 4; ++tn)
#pragma unroll
        for (int j = 0; j < 8; ++j) {
          int a = tm * 16 + j + halfb, n = tn * 16 + colb;
          float v = sc[tm][tn][j];
          if (s_mask[a * SM + n]) v = 0.f;
          s_w[hh * 32 * SW + a * SW + n] = (bf16)v;
        }
  }
  __syncthreads();
  {
    const int hh = wave;
#pragma unroll
    for (int tm = 0; tm < 2; ++tm)
#pragma unroll
      for (int tn = 0; tn < 2; ++tn) {
        v8f acc = zero8();
        const bf16* ab = s_w + hh * 32 * SW + tm * 16 * SW;
        const bf16* bb = s_vT + (hh * 32 + tn * 16) * SV;
#pragma unroll
        for (int kc = 0; kc < 2; ++kc)
          acc = wmma_bf16(load_a(ab, SW, kc * 32, lane), load_b(bb, SV, kc * 32, lane), acc);
        int ec = hh * 32 + tn * 16 + colb;
        int rb = tm * 16 + halfb;
#pragma unroll
        for (int j = 0; j < 8; ++j) s_att[(rb + j) * SH + ec] = (bf16)acc[j];
      }
  }
  __syncthreads();

  // ---------- Phase 4: GRU cell, each wave owns a 32-wide gate-column slice ----------
  {
    v8f ar[2][2], az[2][2], ai[2][2], ah[2][2];
#pragma unroll
    for (int tm = 0; tm < 2; ++tm)
#pragma unroll
      for (int tl = 0; tl < 2; ++tl) {
        ar[tm][tl] = zero8(); az[tm][tl] = zero8();
        ai[tm][tl] = zero8(); ah[tm][tl] = zero8();
      }
    const int c0w = wave * 32;
    for (int kc = 0; kc < 8; ++kc) {
      v16bf aat0 = load_a(s_att,           SH, kc * 32, lane);
      v16bf aat1 = load_a(s_att + 16 * SH, SH, kc * 32, lane);
      v16bf ah00 = load_a(s_h0,            SH, kc * 32, lane);
      v16bf ah01 = load_a(s_h0 + 16 * SH,  SH, kc * 32, lane);
#pragma unroll
      for (int tl = 0; tl < 2; ++tl) {
        int c0 = c0w + tl * 16;
        v16bf br = load_b(wih + (size_t)(c0)*HID,        HID, kc * 32, lane);
        v16bf bz = load_b(wih + (size_t)(256 + c0)*HID,  HID, kc * 32, lane);
        v16bf bi = load_b(wih + (size_t)(512 + c0)*HID,  HID, kc * 32, lane);
        v16bf cr = load_b(whh + (size_t)(c0)*HID,        HID, kc * 32, lane);
        v16bf cz = load_b(whh + (size_t)(256 + c0)*HID,  HID, kc * 32, lane);
        v16bf cn = load_b(whh + (size_t)(512 + c0)*HID,  HID, kc * 32, lane);
        ar[0][tl] = wmma_bf16(aat0, br, ar[0][tl]);
        ar[0][tl] = wmma_bf16(ah00, cr, ar[0][tl]);
        ar[1][tl] = wmma_bf16(aat1, br, ar[1][tl]);
        ar[1][tl] = wmma_bf16(ah01, cr, ar[1][tl]);
        az[0][tl] = wmma_bf16(aat0, bz, az[0][tl]);
        az[0][tl] = wmma_bf16(ah00, cz, az[0][tl]);
        az[1][tl] = wmma_bf16(aat1, bz, az[1][tl]);
        az[1][tl] = wmma_bf16(ah01, cz, az[1][tl]);
        ai[0][tl] = wmma_bf16(aat0, bi, ai[0][tl]);
        ai[1][tl] = wmma_bf16(aat1, bi, ai[1][tl]);
        ah[0][tl] = wmma_bf16(ah00, cn, ah[0][tl]);
        ah[1][tl] = wmma_bf16(ah01, cn, ah[1][tl]);
      }
    }
    const float* gh0 = g_hidden + (size_t)b * AA * HID;
    float* outh = g_out + (size_t)B_TOT * AA * NACT;
#pragma unroll
    for (int tm = 0; tm < 2; ++tm)
#pragma unroll
      for (int tl = 0; tl < 2; ++tl) {
        int c = c0w + tl * 16 + colb;
        float b_r = b_ih[c] + b_hh[c];
        float b_z = b_ih[256 + c] + b_hh[256 + c];
        float b_i = b_ih[512 + c];
        float b_n = b_hh[512 + c];
#pragma unroll
        for (int j = 0; j < 8; ++j) {
          int a = tm * 16 + j + halfb;
          float r  = sigmoidf_(ar[tm][tl][j] + b_r);
          float z  = sigmoidf_(az[tm][tl][j] + b_z);
          float nn = tanhf(ai[tm][tl][j] + b_i + r * (ah[tm][tl][j] + b_n));
          float h0v = gh0[a * HID + c];
          float hv = (1.0f - z) * nn + z * h0v;
          if (g_scen[(size_t)b * AA + a]) hv = 0.f;
          outh[((size_t)b * AA + a) * HID + c] = hv;
          s_h[a * SH + c] = (bf16)hv;
        }
      }
  }
  __syncthreads();

  // ---------- Phase 5: q_out = h @ out_w^T + out_b (32x32, K=256) ----------
  if (wave < 4) {
    int tm = wave >> 1, tn = wave & 1;
    v8f acc = zero8();
    const bf16* ab = s_h + tm * 16 * SH;
    const bf16* bb = wout + tn * 16 * HID;
#pragma unroll
    for (int kc = 0; kc < 8; ++kc)
      acc = wmma_bf16(load_a(ab, SH, kc * 32, lane), load_b(bb, HID, kc * 32, lane), acc);
    int c = tn * 16 + colb;
    float bo = out_b[c];
#pragma unroll
    for (int j = 0; j < 8; ++j) {
      int a = tm * 16 + j + halfb;
      float qv = acc[j] + bo;
      if (g_scen[(size_t)b * AA + a]) qv = 0.f;
      g_out[((size_t)b * AA + a) * NACT + c] = qv;
    }
  }
}

extern "C" void kernel_launch(void* const* d_in, const int* in_sizes, int n_in,
                              void* d_out, int out_size, void* d_ws, size_t ws_size,
                              hipStream_t stream) {
  const float* inputs  = (const float*)d_in[0];
  const float* hidden  = (const float*)d_in[1];
  const float* fc1_w   = (const float*)d_in[2];
  const float* fc1_b   = (const float*)d_in[3];
  const float* q_w     = (const float*)d_in[4];
  const float* q_b     = (const float*)d_in[5];
  const float* k_w     = (const float*)d_in[6];
  const float* k_b     = (const float*)d_in[7];
  const float* v_w     = (const float*)d_in[8];
  const float* v_b     = (const float*)d_in[9];
  const float* w_ih    = (const float*)d_in[10];
  const float* w_hh    = (const float*)d_in[11];
  const float* b_ih    = (const float*)d_in[12];
  const float* b_hh    = (const float*)d_in[13];
  const float* out_w   = (const float*)d_in[14];
  const float* out_b   = (const float*)d_in[15];
  const unsigned char* obs_mask = (const unsigned char*)d_in[16];
  const unsigned char* scen     = (const unsigned char*)d_in[17];
  bf16* ws = (bf16*)d_ws;

  convert_weights_kernel<<<(WS_TOTAL + 255) / 256, 256, 0, stream>>>(
      fc1_w, q_w, k_w, v_w, w_ih, w_hh, out_w, ws);
  attn_rnn_agent_kernel<<<B_TOT, 256, 0, stream>>>(
      inputs, hidden, fc1_b, q_b, k_b, v_b, b_ih, b_hh, out_b,
      obs_mask, scen, ws, (float*)d_out);
}
